// SwinTransformerBlock_18021682774778
// MI455X (gfx1250) — matrix operand order repaired and checked by reading, not compiled, not run on hardware
//
#include <hip/hip_runtime.h>
#include <math.h>

typedef __bf16 bf16;
typedef __attribute__((ext_vector_type(16))) __bf16 v16bf;
typedef __attribute__((ext_vector_type(8)))  __bf16 v8bf;
typedef __attribute__((ext_vector_type(8)))  float   v8f;

#define TKN    200704   // B*H*W tokens
#define CDIM   192
#define NHEAD  6
#define HDIM   32
#define NWIN   4096     // B * (56/7)^2 windows
#define NTOK   49       // tokens per 7x7 window
#define MTILES 12544    // TKN / 16
#define HIDD   768
#define QK_SCALE 0.17677669529663687f  // 32^-0.5

// ---------- helpers ----------

static __device__ __forceinline__ bf16 f2bf(float f) {
    union { float f; unsigned u; } x; x.f = f;
    unsigned r = (x.u + 0x7FFFu + ((x.u >> 16) & 1u)) >> 16;
    union { unsigned short s; bf16 b; } y; y.s = (unsigned short)r;
    return y.b;
}

// A-fragment (16x32 bf16): lane l holds row m=l&15;
// elems 0..15 -> K = hi*16 + i  (hi = l>>4)
static __device__ __forceinline__ v16bf load_a_frag(const bf16* A, int lda, int row0,
                                                    int k0, int lane) {
    int m  = lane & 15;
    int hi = lane >> 4;
    const bf16* p = A + (row0 + m) * lda + k0 + hi * 16;   // 32-bit index math
    v8bf lo = *(const v8bf*)p;
    v8bf h8 = *(const v8bf*)(p + 8);
    return __builtin_shufflevector(lo, h8, 0,1,2,3,4,5,6,7,8,9,10,11,12,13,14,15);
}

// K-loop producing TWO independent 16x16 accumulators (adjacent n-tiles) that share
// one A-fragment: the two WMMAs per step are independent, so the compiler can fill
// the WMMA->WMMA RAW hazard slots with real work instead of v_nop.
static __device__ __forceinline__ void gemm_acc2(const bf16* __restrict__ A, int lda, int row0,
                                                 const bf16* __restrict__ Wp, int ntp, int KT,
                                                 int lane, v8f& acc0, v8f& acc1) {
    const bf16* wp0 = Wp + (2 * ntp) * KT * 512 + lane * 16;
    const bf16* wp1 = wp0 + KT * 512;
    for (int kt = 0; kt < KT; ++kt) {
        v16bf a  = load_a_frag(A, lda, row0, kt * 32, lane);
        v16bf b0 = *(const v16bf*)wp0; wp0 += 512;
        v16bf b1 = *(const v16bf*)wp1; wp1 += 512;
        acc0 = __builtin_amdgcn_wmma_f32_16x16x32_bf16(false, a, false, b0, (short)0, acc0,
                                                       false, false);
        acc1 = __builtin_amdgcn_wmma_f32_16x16x32_bf16(false, a, false, b1, (short)0, acc1,
                                                       false, false);
    }
}

// ---------- weight repack: fp32 (K x N) row-major -> bf16 WMMA B-fragment layout ----------

__global__ void pack_w_kernel(const float* __restrict__ W, bf16* __restrict__ Wp,
                              int K, int Nn) {
    int idx = blockIdx.x * 256 + threadIdx.x;
    if (idx >= K * Nn) return;
    int KT   = K >> 5;
    int i    = idx & 15;
    int lane = (idx >> 4) & 31;
    int blk  = idx >> 9;
    int kt   = blk % KT;
    int nt   = blk / KT;
    int kk   = i + ((lane >> 4) << 4);      // k within 32-slice
    int n    = nt * 16 + (lane & 15);
    int k    = kt * 32 + kk;
    Wp[idx] = f2bf(W[k * Nn + n]);
}

// ---------- LN1 + cyclic shift + window partition (wave per token) ----------

__global__ void ln1_shift_kernel(const float* __restrict__ x, const float* __restrict__ g,
                                 const float* __restrict__ b, bf16* __restrict__ outw) {
    int wave = (blockIdx.x * blockDim.x + threadIdx.x) >> 5;
    int lane = threadIdx.x & 31;
    unsigned w = (unsigned)wave / NTOK;
    int n = wave - (int)w * NTOK;
    int bb = w >> 6, wi = w & 63;
    int wy = wi >> 3, wx = wi & 7;
    int py = n / 7,  px = n - py * 7;
    int sy = wy * 7 + py + 3; if (sy >= 56) sy -= 56;   // roll(-3)
    int sx = wx * 7 + px + 3; if (sx >= 56) sx -= 56;
    const float* src = x + ((bb * 56 + sy) * 56 + sx) * CDIM;
    float v[6]; float s = 0.f;
#pragma unroll
    for (int r = 0; r < 6; ++r) { v[r] = src[lane + 32 * r]; s += v[r]; }
#pragma unroll
    for (int o = 16; o > 0; o >>= 1) s += __shfl_xor(s, o, 32);
    float mean = s * (1.0f / 192.0f);
    float q = 0.f;
#pragma unroll
    for (int r = 0; r < 6; ++r) { float d = v[r] - mean; q += d * d; }
#pragma unroll
    for (int o = 16; o > 0; o >>= 1) q += __shfl_xor(q, o, 32);
    float rstd = rsqrtf(q * (1.0f / 192.0f) + 1e-5f);
    bf16* dst = outw + wave * CDIM;
#pragma unroll
    for (int r = 0; r < 6; ++r) {
        int c = lane + 32 * r;
        dst[c] = f2bf((v[r] - mean) * rstd * g[c] + b[c]);
    }
}

// ---------- plain LN (natural token order) ----------

__global__ void ln2_kernel(const float* __restrict__ x, const float* __restrict__ g,
                           const float* __restrict__ b, bf16* __restrict__ outw) {
    int wave = (blockIdx.x * blockDim.x + threadIdx.x) >> 5;
    int lane = threadIdx.x & 31;
    const float* src = x + wave * CDIM;
    float v[6]; float s = 0.f;
#pragma unroll
    for (int r = 0; r < 6; ++r) { v[r] = src[lane + 32 * r]; s += v[r]; }
#pragma unroll
    for (int o = 16; o > 0; o >>= 1) s += __shfl_xor(s, o, 32);
    float mean = s * (1.0f / 192.0f);
    float q = 0.f;
#pragma unroll
    for (int r = 0; r < 6; ++r) { float d = v[r] - mean; q += d * d; }
#pragma unroll
    for (int o = 16; o > 0; o >>= 1) q += __shfl_xor(q, o, 32);
    float rstd = rsqrtf(q * (1.0f / 192.0f) + 1e-5f);
    bf16* dst = outw + wave * CDIM;
#pragma unroll
    for (int r = 0; r < 6; ++r) {
        int c = lane + 32 * r;
        dst[c] = f2bf((v[r] - mean) * rstd * g[c] + b[c]);
    }
}

// ---------- QKV GEMM: (T x 192) @ (192 x 576), scatter to per-(win,head) q/k/v ----------

__global__ void qkv_gemm_kernel(const bf16* __restrict__ A, const bf16* __restrict__ Wp,
                                const float* __restrict__ bias, bf16* __restrict__ qkv) {
    int lane = threadIdx.x & 31;
    int wave = (blockIdx.x * blockDim.x + threadIdx.x) >> 5;
    int ntp  = wave % 18;                 // pair of n-tiles (36 tiles total)
    int row0 = (wave / 18) * 16;
    v8f acc0 = {}, acc1 = {};
    gemm_acc2(A, CDIM, row0, Wp, ntp, 6, lane, acc0, acc1);
    int n0 = lane & 15, mb = (lane >> 4) * 8;
#pragma unroll
    for (int half = 0; half < 2; ++half) {
        v8f acc = half ? acc1 : acc0;
        int col = ntp * 32 + half * 16 + n0;
        int which = col / CDIM;
        int c     = col - which * CDIM;
        float bcol = bias[col];
        float scl  = (which == 0) ? QK_SCALE : 1.0f;
        int headoff = ((which * NWIN) * NHEAD + (c >> 5)) * (NTOK * HDIM) + (c & 31);
#pragma unroll
        for (int vv = 0; vv < 8; ++vv) {
            int row = row0 + mb + vv;
            unsigned w = (unsigned)row / NTOK;
            int n = row - (int)w * NTOK;
            float val = (acc[vv] + bcol) * scl;
            int offd = headoff + (int)w * (NHEAD * NTOK * HDIM) + n * HDIM;
            qkv[offd] = f2bf(val);
        }
    }
}

// ---------- attention: one workgroup (4 waves) per (window, head) ----------

__global__ void attn_kernel(const bf16* __restrict__ qkv, const float* __restrict__ rpbt,
                            bf16* __restrict__ ctx) {
    __shared__ float S[64 * 64];
    __shared__ bf16  P[64 * 64];
    int lane = threadIdx.x & 31;
    int wv   = threadIdx.x >> 5;
    int wh   = blockIdx.x;
    int w = wh / NHEAD, h = wh - w * NHEAD;
    const int hdsz = NWIN * NHEAD * NTOK * HDIM;
    const bf16* qb = qkv + (w * NHEAD + h) * (NTOK * HDIM);
    const bf16* kb = qb + hdsz;
    const bf16* vb = qb + 2 * hdsz;

    // S = (Q*scale) @ K^T : wave wv does rows [16wv,16wv+16), 4 independent accumulators
    {
        int m = lane & 15, hi = lane >> 4;
        int rq = wv * 16 + m; if (rq >= NTOK) rq = 0;   // clamp; garbage rows never stored
        const bf16* p = qb + rq * HDIM + hi * 16;
        v8bf lo = *(const v8bf*)p;
        v8bf h8 = *(const v8bf*)(p + 8);
        v16bf afrag = __builtin_shufflevector(lo, h8, 0,1,2,3,4,5,6,7,8,9,10,11,12,13,14,15);
#pragma unroll
        for (int nt = 0; nt < 4; ++nt) {
            int rk = nt * 16 + (lane & 15); if (rk >= NTOK) rk = 0;  // cols>=49 masked later
            v16bf bfrag = *(const v16bf*)(kb + rk * HDIM + hi * 16);
            v8f acc = {};
            acc = __builtin_amdgcn_wmma_f32_16x16x32_bf16(false, afrag, false, bfrag,
                                                          (short)0, acc, false, false);
            int n0 = lane & 15, mbase = hi * 8;
#pragma unroll
            for (int vv = 0; vv < 8; ++vv)
                S[(wv * 16 + mbase + vv) * 64 + nt * 16 + n0] = acc[vv];
        }
    }
    __syncthreads();

    // rel-pos bias + shift mask + softmax; thread per row; probs -> bf16 P (K padded to 64)
    {
        int m = threadIdx.x;
        if (m < 64) {
            if (m < NTOK) {
                int wi = w & 63, wy = wi >> 3, wx = wi & 7;
                int ym = m / 7, xm = m - ym * 7;
                int gy = wy * 7 + ym, gx = wx * 7 + xm;
                int rm = (gy < 49 ? 0 : (gy < 53 ? 1 : 2)) * 3 +
                         (gx < 49 ? 0 : (gx < 53 ? 1 : 2));
                float mx = -1e30f;
                for (int nn = 0; nn < NTOK; ++nn) {
                    int yn = nn / 7, xn = nn - yn * 7;
                    int gyn = wy * 7 + yn, gxn = wx * 7 + xn;
                    int rn = (gyn < 49 ? 0 : (gyn < 53 ? 1 : 2)) * 3 +
                             (gxn < 49 ? 0 : (gxn < 53 ? 1 : 2));
                    int idx = (ym - yn + 6) * 13 + (xm - xn + 6);
                    float sv = S[m * 64 + nn] + rpbt[idx * NHEAD + h] +
                               ((rm == rn) ? 0.0f : -100.0f);
                    S[m * 64 + nn] = sv;
                    mx = fmaxf(mx, sv);
                }
                float sum = 0.0f;
                for (int nn = 0; nn < NTOK; ++nn) {
                    float e = __expf(S[m * 64 + nn] - mx);
                    S[m * 64 + nn] = e;
                    sum += e;
                }
                float inv = 1.0f / sum;
                for (int nn = 0; nn < NTOK; ++nn) P[m * 64 + nn] = f2bf(S[m * 64 + nn] * inv);
                for (int nn = NTOK; nn < 64; ++nn) P[m * 64 + nn] = f2bf(0.0f);
            } else {
                for (int nn = 0; nn < 64; ++nn) P[m * 64 + nn] = f2bf(0.0f);
            }
        }
    }
    __syncthreads();

    // O = P @ V (M=64,K=64,N=32): shared A-fragment, two independent accumulators
    {
        v8f acc0 = {}, acc1 = {};
#pragma unroll
        for (int ks = 0; ks < 2; ++ks) {
            v16bf afrag = load_a_frag(P, 64, wv * 16, ks * 32, lane);   // LDS ds_load path
            int jd = lane & 15, hi = lane >> 4;
            v16bf bfrag0, bfrag1;
#pragma unroll
            for (int i = 0; i < 16; ++i) {
                int r = ks * 32 + hi * 16 + i;
                bf16 z = f2bf(0.0f);
                bfrag0[i] = (r < NTOK) ? vb[r * HDIM + jd]      : z;
                bfrag1[i] = (r < NTOK) ? vb[r * HDIM + 16 + jd] : z;
            }
            acc0 = __builtin_amdgcn_wmma_f32_16x16x32_bf16(false, afrag, false, bfrag0,
                                                           (short)0, acc0, false, false);
            acc1 = __builtin_amdgcn_wmma_f32_16x16x32_bf16(false, afrag, false, bfrag1,
                                                           (short)0, acc1, false, false);
        }
        int n0 = lane & 15, mbase = (lane >> 4) * 8;
#pragma unroll
        for (int vv = 0; vv < 8; ++vv) {
            int mrow = wv * 16 + mbase + vv;
            if (mrow < NTOK) {
                int base = (w * NTOK + mrow) * CDIM + h * HDIM + n0;
                ctx[base]      = f2bf(acc0[vv]);
                ctx[base + 16] = f2bf(acc1[vv]);
            }
        }
    }
}

// ---------- proj GEMM + window reverse + un-roll + residual -> x1 (natural order) ----------

__global__ void proj_gemm_kernel(const bf16* __restrict__ A, const bf16* __restrict__ Wp,
                                 const float* __restrict__ bias, const float* __restrict__ xin,
                                 float* __restrict__ x1) {
    int lane = threadIdx.x & 31;
    int wave = (blockIdx.x * blockDim.x + threadIdx.x) >> 5;
    int ntp  = wave % 6;                  // 12 n-tiles -> 6 pairs
    int row0 = (wave / 6) * 16;
    v8f acc0 = {}, acc1 = {};
    gemm_acc2(A, CDIM, row0, Wp, ntp, 6, lane, acc0, acc1);
    int n0 = lane & 15, mb = (lane >> 4) * 8;
#pragma unroll
    for (int vv = 0; vv < 8; ++vv) {
        int row = row0 + mb + vv;
        unsigned w = (unsigned)row / NTOK;
        int n = row - (int)w * NTOK;
        int bb = (int)(w >> 6), wi = (int)(w & 63);
        int wy = wi >> 3, wx = wi & 7;
        int py = n / 7, px = n - py * 7;
        int hy = wy * 7 + py + 3; if (hy >= 56) hy -= 56;   // roll(+3)
        int hx = wx * 7 + px + 3; if (hx >= 56) hx -= 56;
        int t = (bb * 56 + hy) * 56 + hx;
        int col0 = ntp * 32 + n0;
        x1[t * CDIM + col0]      = xin[t * CDIM + col0]      + acc0[vv] + bias[col0];
        x1[t * CDIM + col0 + 16] = xin[t * CDIM + col0 + 16] + acc1[vv] + bias[col0 + 16];
    }
}

// ---------- FC1 GEMM + exact GELU ----------

__global__ void fc1_gemm_kernel(const bf16* __restrict__ A, const bf16* __restrict__ Wp,
                                const float* __restrict__ bias, bf16* __restrict__ hb) {
    int lane = threadIdx.x & 31;
    int wave = (blockIdx.x * blockDim.x + threadIdx.x) >> 5;
    int ntp  = wave % 24;                 // 48 n-tiles -> 24 pairs
    int row0 = (wave / 24) * 16;
    v8f acc0 = {}, acc1 = {};
    gemm_acc2(A, CDIM, row0, Wp, ntp, 6, lane, acc0, acc1);
    int n0 = lane & 15, mb = (lane >> 4) * 8;
    int col0 = ntp * 32 + n0;
    float b0 = bias[col0], b1 = bias[col0 + 16];
#pragma unroll
    for (int vv = 0; vv < 8; ++vv) {
        int row = row0 + mb + vv;
        float v0 = acc0[vv] + b0;
        float v1 = acc1[vv] + b1;
        float g0 = 0.5f * v0 * (1.0f + erff(v0 * 0.70710678118654752f));
        float g1 = 0.5f * v1 * (1.0f + erff(v1 * 0.70710678118654752f));
        hb[row * HIDD + col0]      = f2bf(g0);
        hb[row * HIDD + col0 + 16] = f2bf(g1);
    }
}

// ---------- FC2 GEMM + residual -> final output ----------

__global__ void fc2_gemm_kernel(const bf16* __restrict__ A, const bf16* __restrict__ Wp,
                                const float* __restrict__ bias, const float* __restrict__ x1,
                                float* __restrict__ out) {
    int lane = threadIdx.x & 31;
    int wave = (blockIdx.x * blockDim.x + threadIdx.x) >> 5;
    int ntp  = wave % 6;                  // 12 n-tiles -> 6 pairs
    int row0 = (wave / 6) * 16;
    v8f acc0 = {}, acc1 = {};
    gemm_acc2(A, HIDD, row0, Wp, ntp, 24, lane, acc0, acc1);
    int n0 = lane & 15, mb = (lane >> 4) * 8;
    int col0 = ntp * 32 + n0;
    float b0 = bias[col0], b1 = bias[col0 + 16];
#pragma unroll
    for (int vv = 0; vv < 8; ++vv) {
        int row = row0 + mb + vv;
        out[row * CDIM + col0]      = x1[row * CDIM + col0]      + acc0[vv] + b0;
        out[row * CDIM + col0 + 16] = x1[row * CDIM + col0 + 16] + acc1[vv] + b1;
    }
}

// ---------- host launcher ----------

extern "C" void kernel_launch(void* const* d_in, const int* in_sizes, int n_in,
                              void* d_out, int out_size, void* d_ws, size_t ws_size,
                              hipStream_t stream) {
    (void)in_sizes; (void)n_in; (void)out_size; (void)ws_size;
    const float* x     = (const float*)d_in[0];
    const float* n1g   = (const float*)d_in[1];
    const float* n1b   = (const float*)d_in[2];
    const float* qkvw  = (const float*)d_in[3];
    const float* qkvb  = (const float*)d_in[4];
    const float* rpbt  = (const float*)d_in[5];
    const float* projw = (const float*)d_in[6];
    const float* projb = (const float*)d_in[7];
    const float* n2g   = (const float*)d_in[8];
    const float* n2b   = (const float*)d_in[9];
    const float* fc1w  = (const float*)d_in[10];
    const float* fc1b  = (const float*)d_in[11];
    const float* fc2w  = (const float*)d_in[12];
    const float* fc2b  = (const float*)d_in[13];
    float* out = (float*)d_out;

    char* ws = (char*)d_ws;
    size_t off = 0;
    auto take = [&](size_t bytes) -> char* {
        char* p = ws + off;
        off += (bytes + 255) & ~(size_t)255;
        return p;
    };
    bf16*  xw    = (bf16*)take((size_t)TKN * CDIM * 2);            // LN1+shift (bf16)
    bf16*  qkvB  = (bf16*)take((size_t)3 * TKN * CDIM * 2);        // q,k,v per (win,head)
    bf16*  ctx   = (bf16*)take((size_t)TKN * CDIM * 2);            // attention output
    float* x1    = (float*)take((size_t)TKN * CDIM * 4);           // residual 1
    bf16*  xh    = (bf16*)take((size_t)TKN * CDIM * 2);            // LN2 out
    bf16*  hb    = (bf16*)take((size_t)TKN * HIDD * 2);            // FC1 out
    bf16*  wqkv  = (bf16*)take((size_t)192 * 576 * 2);
    bf16*  wproj = (bf16*)take((size_t)192 * 192 * 2);
    bf16*  wfc1  = (bf16*)take((size_t)192 * 768 * 2);
    bf16*  wfc2  = (bf16*)take((size_t)768 * 192 * 2);

    pack_w_kernel<<<(192 * 576 + 255) / 256, 256, 0, stream>>>(qkvw,  wqkv,  192, 576);
    pack_w_kernel<<<(192 * 192 + 255) / 256, 256, 0, stream>>>(projw, wproj, 192, 192);
    pack_w_kernel<<<(192 * 768 + 255) / 256, 256, 0, stream>>>(fc1w,  wfc1,  192, 768);
    pack_w_kernel<<<(768 * 192 + 255) / 256, 256, 0, stream>>>(fc2w,  wfc2,  768, 192);

    ln1_shift_kernel<<<TKN / 8, 256, 0, stream>>>(x, n1g, n1b, xw);
    qkv_gemm_kernel<<<(MTILES * 18) / 8, 256, 0, stream>>>(xw, wqkv, qkvb, qkvB);
    attn_kernel<<<NWIN * NHEAD, 128, 0, stream>>>(qkvB, rpbt, ctx);
    proj_gemm_kernel<<<(MTILES * 6) / 8, 256, 0, stream>>>(ctx, wproj, projb, x, x1);
    ln2_kernel<<<TKN / 8, 256, 0, stream>>>(x1, n2g, n2b, xh);
    fc1_gemm_kernel<<<(MTILES * 24) / 8, 256, 0, stream>>>(xh, wfc1, fc1b, hb);
    fc2_gemm_kernel<<<(MTILES * 6) / 8, 256, 0, stream>>>(hb, wfc2, fc2b, x1, out);
}
